// TraitsPredictor_77644418777464
// MI455X (gfx1250) — matrix operand order repaired and checked by reading, not compiled, not run on hardware
//
#include <hip/hip_runtime.h>
#include <math.h>

// ---------------- problem constants ----------------
#define N_SP_   20000
#define N_SPAT_ 10000
#define E_SS_   160000
#define E_BIP_  320000
#define E_SP_   320000
#define HEADS_  4
#define CH_     64

typedef __attribute__((ext_vector_type(16))) _Float16 v16h;
typedef __attribute__((ext_vector_type(8)))  _Float16 v8h;
typedef __attribute__((ext_vector_type(8)))  float    v8f;

static inline unsigned gsz(long n, int b) { return (unsigned)((n + (long)b - 1) / b); }

// =====================================================================
// GEMM prep: f32 -> f16 with zero-padding of K to Kp (multiple of 32)
// =====================================================================
// activations: src f32 [M,K] -> dst f16 [M,Kp]
__global__ void cvt_a_f16(const float* __restrict__ src, _Float16* __restrict__ dst,
                          int M, int K, int Kp) {
    long i = blockIdx.x * (long)blockDim.x + threadIdx.x;
    if (i >= (long)M * Kp) return;
    int r = (int)(i / Kp), k = (int)(i % Kp);
    dst[i] = (k < K) ? (_Float16)src[(long)r * K + k] : (_Float16)0.f;
}

// weights: W f32 [K,N] -> Wt f16 [N,Kp] (transposed)
__global__ void cvt_w_f16t(const float* __restrict__ W, _Float16* __restrict__ Wt,
                           int K, int N, int Kp) {
    long i = blockIdx.x * (long)blockDim.x + threadIdx.x;
    if (i >= (long)N * Kp) return;
    int c = (int)(i / Kp), k = (int)(i % Kp);
    Wt[i] = (k < K) ? (_Float16)W[(long)k * N + c] : (_Float16)0.f;
}

// =====================================================================
// WMMA GEMM: out[M,N] = act(Ah[M,Kp] @ Wt[N,Kp]^T + bias)
// block = 128 threads = 4 wave32; block tile 16x64; per-wave 16x16 tile.
// Fragments are loaded DIRECTLY from global memory (no LDS): with the
// f16 [.,Kp] layouts, each lane's fragment (k = 16*half + s, same for A and B) is contiguous:
//   A lane = m + 16*half : halves [k0+half*16, +8) and [k0+half*16+8, +8)
//   B lane = (c&15)+16*kh: halves [k0+kh*16, +16)
// =====================================================================
__launch_bounds__(128)
__global__ void wmma_gemm_k(const _Float16* __restrict__ Ah, const _Float16* __restrict__ Wt,
                            const float* __restrict__ bias, float* __restrict__ out,
                            int M, int Kp, int N, int do_relu)
{
    const int tid  = threadIdx.x;
    const int wave = tid >> 5;
    const int lane = tid & 31;
    const int half = lane >> 4;
    const int l16  = lane & 15;
    const int row0 = blockIdx.y * 16;
    const int col0 = blockIdx.x * 64;

    int ar = row0 + l16;          if (ar >= M) ar = M - 1;   // clamp (stores are guarded)
    const int gc = col0 + wave * 16 + l16;
    int bc = gc;                  if (bc >= N) bc = N - 1;

    const _Float16* aptr = Ah + (long)ar * Kp + half * 16;
    const _Float16* bptr = Wt + (long)bc * Kp + half * 16;

    v8f acc = {};
    for (int k0 = 0; k0 < Kp; k0 += 32) {
        __builtin_prefetch(aptr + k0 + 32, 0, 1);   // global_prefetch_b8
        __builtin_prefetch(bptr + k0 + 32, 0, 1);
        v8h alo = *(const v8h*)(aptr + k0);
        v8h ahi = *(const v8h*)(aptr + k0 + 8);
        v16h af = __builtin_shufflevector(alo, ahi, 0, 1, 2, 3, 4, 5, 6, 7,
                                          8, 9, 10, 11, 12, 13, 14, 15);
        v16h bf = *(const v16h*)(bptr + k0);
        acc = __builtin_amdgcn_wmma_f32_16x16x32_f16(false, af, false, bf,
                                                     (short)0, acc, false, false);
    }

    // store: C/D layout -> VGPR v holds M=v (lanes 0-15) / M=v+8 (lanes 16-31)
    if (gc < N) {
        float bb = bias ? bias[gc] : 0.f;
#pragma unroll
        for (int v = 0; v < 8; ++v) {
            int gr = row0 + v + 8 * half;
            if (gr < M) {
                float val = acc[v] + bb;
                if (do_relu) val = val > 0.f ? val : 0.f;
                out[(long)gr * N + gc] = val;
            }
        }
    }
}

struct GemmScratch { _Float16* Ah; _Float16* Wt; };

static void gemm(hipStream_t st, const GemmScratch& G, const float* A, const float* W,
                 const float* bias, float* out, int M, int K, int N, int do_relu)
{
    const int Kp = (K + 31) & ~31;
    cvt_a_f16 <<<gsz((long)M * Kp, 256), 256, 0, st>>>(A, G.Ah, M, K, Kp);
    cvt_w_f16t<<<gsz((long)N * Kp, 256), 256, 0, st>>>(W, G.Wt, K, N, Kp);
    dim3 g((N + 63) / 64, (M + 15) / 16);
    wmma_gemm_k<<<g, 128, 0, st>>>(G.Ah, G.Wt, bias, out, M, Kp, N, do_relu);
}

// =====================================================================
// small helper kernels
// =====================================================================
__global__ void fill_f32(float* p, float v, long n) {
    long i = blockIdx.x * (long)blockDim.x + threadIdx.x;
    if (i < n) p[i] = v;
}

__global__ void pack_cols(float* dst, int ldd, int col0, const float* __restrict__ src,
                          int ncols, int M) {
    long i = blockIdx.x * (long)blockDim.x + threadIdx.x;
    long tot = (long)M * ncols;
    if (i >= tot) return;
    int r = (int)(i / ncols), c = (int)(i % ncols);
    dst[(long)r * ldd + col0 + c] = src[i];
}

__global__ void pack_vis(float* dst, int ldd, int col0, const unsigned char* __restrict__ mask,
                         int ncols, int M) {
    long i = blockIdx.x * (long)blockDim.x + threadIdx.x;
    long tot = (long)M * ncols;
    if (i >= tot) return;
    int r = (int)(i / ncols), c = (int)(i % ncols);
    dst[(long)r * ldd + col0 + c] = mask[i] ? 0.f : 1.f;   // vis = ~nanmask
}

// al[n,h] = sum_c hs[n,h,c] * a[h,c]
__global__ void node_logits(const float* __restrict__ hs, const float* __restrict__ a,
                            float* __restrict__ al, int N, int H, int C) {
    long i = blockIdx.x * (long)blockDim.x + threadIdx.x;
    if (i >= (long)N * H) return;
    int n = (int)(i / H), h = (int)(i % H);
    const float* hp = hs + (long)n * H * C + (long)h * C;
    const float* ap = a + (long)h * C;
    float s = 0.f;
    for (int c = 0; c < C; ++c) s += hp[c] * ap[c];
    al[i] = s;
}

// wedot[h] = sum_c We[h*C+c] * a_e[h*C+c]   (edge_dim == 1)
__global__ void compute_wedot(const float* __restrict__ We, const float* __restrict__ ae,
                              float* __restrict__ wedot, int H, int C) {
    int h = threadIdx.x;
    if (h < H) {
        float s = 0.f;
        for (int c = 0; c < C; ++c) s += We[h * C + c] * ae[h * C + c];
        wedot[h] = s;
    }
}

__global__ void loopea_accum(const int* __restrict__ col, const float* __restrict__ ea,
                             float* cnt, float* easum, int E) {
    int e = blockIdx.x * blockDim.x + threadIdx.x;
    if (e >= E) return;
    atomicAdd(&cnt[col[e]], 1.f);
    atomicAdd(&easum[col[e]], ea[e]);
}

__global__ void loopea_final(float* easum, const float* cnt, int N) {
    int n = blockIdx.x * blockDim.x + threadIdx.x;
    if (n < N) easum[n] = easum[n] / fmaxf(cnt[n], 1.f);
}

__device__ inline void atomicMaxFloat(float* addr, float val) {
    unsigned int* ua = (unsigned int*)addr;
    unsigned int old = *ua;
    while (__uint_as_float(old) < val) {
        unsigned int assumed = old;
        old = atomicCAS(ua, assumed, __float_as_uint(val));
        if (old == assumed) break;
    }
}

__global__ void edge_logits(const int* __restrict__ row, const int* __restrict__ col,
                            const float* __restrict__ ea, const float* __restrict__ als,
                            const float* __restrict__ ald, const float* __restrict__ wedot,
                            float* __restrict__ elog, float* m, int E, int H) {
    long i = blockIdx.x * (long)blockDim.x + threadIdx.x;
    if (i >= (long)E * H) return;
    int e = (int)(i / H), h = (int)(i % H);
    float v = als[(long)row[e] * H + h] + ald[(long)col[e] * H + h] + ea[e] * wedot[h];
    v = v > 0.f ? v : 0.2f * v;  // leaky_relu(0.2)
    elog[i] = v;
    atomicMaxFloat(&m[(long)col[e] * H + h], v);
}

__global__ void self_logits(const float* __restrict__ loop_ea, const float* __restrict__ als,
                            const float* __restrict__ ald, const float* __restrict__ wedot,
                            float* __restrict__ slog, float* m, int N, int H) {
    long i = blockIdx.x * (long)blockDim.x + threadIdx.x;
    if (i >= (long)N * H) return;
    int n = (int)(i / H), h = (int)(i % H);
    float v = als[(long)n * H + h] + ald[(long)n * H + h] + loop_ea[n] * wedot[h];
    v = v > 0.f ? v : 0.2f * v;
    slog[i] = v;
    atomicMaxFloat(&m[i], v);
}

__global__ void edge_exp(const int* __restrict__ col, float* __restrict__ elog,
                         const float* __restrict__ m, float* den, int E, int H) {
    long i = blockIdx.x * (long)blockDim.x + threadIdx.x;
    if (i >= (long)E * H) return;
    int e = (int)(i / H), h = (int)(i % H);
    float ex = expf(elog[i] - m[(long)col[e] * H + h]);
    elog[i] = ex;
    atomicAdd(&den[(long)col[e] * H + h], ex);
}

__global__ void self_exp(float* __restrict__ slog, const float* __restrict__ m,
                         float* den, int N, int H) {
    long i = blockIdx.x * (long)blockDim.x + threadIdx.x;
    if (i >= (long)N * H) return;
    float ex = expf(slog[i] - m[i]);
    slog[i] = ex;
    atomicAdd(&den[i], ex);
}

// one block per edge; blockDim = H*C
__global__ void edge_scatter(const int* __restrict__ row, const int* __restrict__ col,
                             const float* __restrict__ elog, const float* __restrict__ den,
                             const float* __restrict__ hs, float* acc, int H, int C) {
    int e = blockIdx.x;
    int j = threadIdx.x;          // j in [0, H*C)
    int h = j / C;
    int r = row[e], c = col[e];
    float alpha = elog[(long)e * H + h] / (den[(long)c * H + h] + 1e-16f);
    atomicAdd(&acc[(long)c * H * C + j], hs[(long)r * H * C + j] * alpha);
}

// one block per node (self loop); non-atomic (runs after edge_scatter on same stream)
__global__ void self_scatter(const float* __restrict__ slog, const float* __restrict__ den,
                             const float* __restrict__ hs, float* acc, int H, int C) {
    int n = blockIdx.x;
    int j = threadIdx.x;
    int h = j / C;
    float alpha = slog[(long)n * H + h] / (den[(long)n * H + h] + 1e-16f);
    acc[(long)n * H * C + j] += hs[(long)n * H * C + j] * alpha;
}

// out[n,c] = relu( mean_h acc[n,h,c] + b[c] )   (H==1 handles concat case)
__global__ void head_mean_bias_relu(const float* __restrict__ acc, const float* __restrict__ b,
                                    float* __restrict__ out, int N, int H, int C) {
    long i = blockIdx.x * (long)blockDim.x + threadIdx.x;
    if (i >= (long)N * C) return;
    int n = (int)(i / C), c = (int)(i % C);
    float s = 0.f;
    for (int h = 0; h < H; ++h) s += acc[(long)n * H * C + (long)h * C + c];
    s = s / (float)H + b[c];
    out[i] = s > 0.f ? s : 0.f;
}

// ms[N,64] -> mean = ms[:, :32]; std = softplus(ms[:,32:]) + 1e-6
__global__ void split_softplus(const float* __restrict__ ms, float* __restrict__ out, int N) {
    long i = blockIdx.x * (long)blockDim.x + threadIdx.x;
    if (i >= (long)N * 64) return;
    int n = (int)(i >> 6), c = (int)(i & 63);
    float v = ms[i];
    if (c < 32) {
        out[(long)n * 32 + c] = v;
    } else {
        float sp = fmaxf(v, 0.f) + log1pf(expf(-fabsf(v)));   // stable softplus
        out[(long)N * 32 + (long)n * 32 + (c - 32)] = sp + 1e-6f;
    }
}

// =====================================================================
// GAT layer driver (host side, all launches on `st`)
// =====================================================================
struct GatWS {
    float *hs, *hd, *als, *ald, *wedot, *cnt, *easum, *m, *den, *elog, *slog, *acc;
};

static void gat_layer(hipStream_t st, const GatWS& B, const GemmScratch& G,
                      const float* xs, int Ns, int Ks, const float* Ws,
                      const float* xd, int Nd, int Kd, const float* Wd,
                      const int* row, const int* col, const float* ea, int E,
                      const float* a_s, const float* a_d, const float* a_e, const float* We,
                      const float* bias, int H, int C, int self_loops, float* out)
{
    const int HC = H * C;
    // source / dest feature transforms (WMMA GEMMs)
    gemm(st, G, xs, Ws, nullptr, B.hs, Ns, Ks, HC, 0);
    const float* hd = B.hs;
    if (xd != xs || Wd != Ws) {
        gemm(st, G, xd, Wd, nullptr, B.hd, Nd, Kd, HC, 0);
        hd = B.hd;
    }
    node_logits<<<gsz((long)Ns * H, 256), 256, 0, st>>>(B.hs, a_s, B.als, Ns, H, C);
    node_logits<<<gsz((long)Nd * H, 256), 256, 0, st>>>(hd,   a_d, B.ald, Nd, H, C);
    compute_wedot<<<1, 32, 0, st>>>(We, a_e, B.wedot, H, C);

    if (self_loops) {
        fill_f32<<<gsz(Nd, 256), 256, 0, st>>>(B.cnt, 0.f, Nd);
        fill_f32<<<gsz(Nd, 256), 256, 0, st>>>(B.easum, 0.f, Nd);
        loopea_accum<<<gsz(E, 256), 256, 0, st>>>(col, ea, B.cnt, B.easum, E);
        loopea_final<<<gsz(Nd, 256), 256, 0, st>>>(B.easum, B.cnt, Nd);
    }

    fill_f32<<<gsz((long)Nd * H, 256), 256, 0, st>>>(B.m, -3.0e38f, (long)Nd * H);
    edge_logits<<<gsz((long)E * H, 256), 256, 0, st>>>(row, col, ea, B.als, B.ald,
                                                       B.wedot, B.elog, B.m, E, H);
    if (self_loops)
        self_logits<<<gsz((long)Nd * H, 256), 256, 0, st>>>(B.easum, B.als, B.ald,
                                                            B.wedot, B.slog, B.m, Nd, H);

    fill_f32<<<gsz((long)Nd * H, 256), 256, 0, st>>>(B.den, 0.f, (long)Nd * H);
    edge_exp<<<gsz((long)E * H, 256), 256, 0, st>>>(col, B.elog, B.m, B.den, E, H);
    if (self_loops)
        self_exp<<<gsz((long)Nd * H, 256), 256, 0, st>>>(B.slog, B.m, B.den, Nd, H);

    fill_f32<<<gsz((long)Nd * HC, 256), 256, 0, st>>>(B.acc, 0.f, (long)Nd * HC);
    edge_scatter<<<E, HC, 0, st>>>(row, col, B.elog, B.den, B.hs, B.acc, H, C);
    if (self_loops)
        self_scatter<<<Nd, HC, 0, st>>>(B.slog, B.den, B.hs, B.acc, H, C);

    head_mean_bias_relu<<<gsz((long)Nd * C, 256), 256, 0, st>>>(B.acc, bias, out, Nd, H, C);
}

// =====================================================================
// entry point
// =====================================================================
extern "C" void kernel_launch(void* const* d_in, const int* in_sizes, int n_in,
                              void* d_out, int out_size, void* d_ws, size_t ws_size,
                              hipStream_t stream)
{
    (void)in_sizes; (void)n_in; (void)out_size; (void)ws_size;
    hipStream_t st = stream;

    // ---------------- inputs ----------------
    const float* sp_mean   = (const float*)d_in[0];           // [20000,32]
    const float* sp_std    = (const float*)d_in[1];           // [20000,32]
    const unsigned char* nanmask = (const unsigned char*)d_in[2]; // [20000,32] bool
    const float* sp_gen    = (const float*)d_in[3];           // [20000,64]
    const float* sp_phylo  = (const float*)d_in[4];           // [20000,128]
    const float* spat_x    = (const float*)d_in[5];           // [10000,12]
    const float* spat_g    = (const float*)d_in[6];           // [10000,4]
    const int*   ss_ei     = (const int*)d_in[7];             // [2,160000]
    const float* ss_ea     = (const float*)d_in[8];
    const int*   bip_ei    = (const int*)d_in[9];             // [2,320000]
    const float* bip_ea    = (const float*)d_in[10];
    const int*   sp_ei     = (const int*)d_in[11];            // [2,320000]
    const float* sp_ea     = (const float*)d_in[12];

    const float* sg0_W = (const float*)d_in[13];
    const float* sg0_as = (const float*)d_in[14];
    const float* sg0_ad = (const float*)d_in[15];
    const float* sg0_ae = (const float*)d_in[16];
    const float* sg0_We = (const float*)d_in[17];
    const float* sg0_b  = (const float*)d_in[18];
    const float* sg1_W = (const float*)d_in[19];
    const float* sg1_as = (const float*)d_in[20];
    const float* sg1_ad = (const float*)d_in[21];
    const float* sg1_ae = (const float*)d_in[22];
    const float* sg1_We = (const float*)d_in[23];
    const float* sg1_b  = (const float*)d_in[24];
    const float* pg0_W = (const float*)d_in[25];
    const float* pg0_as = (const float*)d_in[26];
    const float* pg0_ad = (const float*)d_in[27];
    const float* pg0_ae = (const float*)d_in[28];
    const float* pg0_We = (const float*)d_in[29];
    const float* pg0_b  = (const float*)d_in[30];
    const float* pg1_W = (const float*)d_in[31];
    const float* pg1_as = (const float*)d_in[32];
    const float* pg1_ad = (const float*)d_in[33];
    const float* pg1_ae = (const float*)d_in[34];
    const float* pg1_We = (const float*)d_in[35];
    const float* pg1_b  = (const float*)d_in[36];
    const float* bp_Ws = (const float*)d_in[37];
    const float* bp_Wd = (const float*)d_in[38];
    const float* bp_as = (const float*)d_in[39];
    const float* bp_ad = (const float*)d_in[40];
    const float* bp_ae = (const float*)d_in[41];
    const float* bp_We = (const float*)d_in[42];
    const float* bp_b  = (const float*)d_in[43];
    const float* sl_W  = (const float*)d_in[44];
    const float* sl_b  = (const float*)d_in[45];
    const float* fc_W  = (const float*)d_in[46];
    const float* fc_b  = (const float*)d_in[47];

    // ---------------- workspace carve (floats, 64B-aligned chunks) ----------------
    float* ws = (float*)d_ws;
    size_t off = 0;
    auto alloc = [&](size_t n) { n = (n + 15) & ~(size_t)15; float* p = ws + off; off += n; return p; };

    float* A288 = alloc((size_t)N_SP_ * 288);
    float* spin = alloc((size_t)N_SP_ * 64);
    float* scat = alloc((size_t)N_SPAT_ * 16);
    float* h1   = alloc((size_t)N_SPAT_ * 64);
    float* h2   = alloc((size_t)N_SPAT_ * 64);

    GatWS B;
    B.hs    = alloc((size_t)N_SP_ * 256);
    B.hd    = alloc((size_t)N_SP_ * 64);
    B.als   = alloc((size_t)N_SP_ * HEADS_);
    B.ald   = alloc((size_t)N_SP_ * HEADS_);
    B.wedot = alloc(16);
    B.cnt   = alloc((size_t)N_SP_);
    B.easum = alloc((size_t)N_SP_);
    B.m     = alloc((size_t)N_SP_ * HEADS_);
    B.den   = alloc((size_t)N_SP_ * HEADS_);
    B.elog  = alloc((size_t)E_SP_ * HEADS_);
    B.slog  = alloc((size_t)N_SP_ * HEADS_);
    B.acc   = alloc((size_t)N_SP_ * 256);

    float* part = alloc((size_t)N_SP_ * 192);
    float* sts  = alloc((size_t)N_SP_ * 64);
    float* xcat = alloc((size_t)N_SP_ * 128);
    float* x1   = alloc((size_t)N_SP_ * 64);
    float* x2   = alloc((size_t)N_SP_ * 64);
    float* msb  = alloc((size_t)N_SP_ * 64);

    GemmScratch G;
    G.Ah = (_Float16*)alloc((size_t)N_SP_ * 288 / 2);   // 5.76M halves
    G.Wt = (_Float16*)alloc(40960);                     // 256*320 halves max

    // ---------------- A) species input MLP ----------------
    pack_cols<<<gsz((long)N_SP_ * 32, 256), 256, 0, st>>>(A288, 288, 0,   sp_mean, 32, N_SP_);
    pack_cols<<<gsz((long)N_SP_ * 32, 256), 256, 0, st>>>(A288, 288, 32,  sp_std,  32, N_SP_);
    pack_vis <<<gsz((long)N_SP_ * 32, 256), 256, 0, st>>>(A288, 288, 64,  nanmask, 32, N_SP_);
    pack_cols<<<gsz((long)N_SP_ * 64, 256), 256, 0, st>>>(A288, 288, 96,  sp_gen,  64, N_SP_);
    pack_cols<<<gsz((long)N_SP_ * 128,256), 256, 0, st>>>(A288, 288, 160, sp_phylo,128, N_SP_);
    gemm(st, G, A288, sl_W, sl_b, spin, N_SP_, 288, 64, 1);   // relu

    // ---------------- B) spatial GNN ----------------
    pack_cols<<<gsz((long)N_SPAT_ * 12, 256), 256, 0, st>>>(scat, 16, 0,  spat_x, 12, N_SPAT_);
    pack_cols<<<gsz((long)N_SPAT_ * 4,  256), 256, 0, st>>>(scat, 16, 12, spat_g, 4,  N_SPAT_);
    const int* ss_row = ss_ei;
    const int* ss_col = ss_ei + E_SS_;
    gat_layer(st, B, G, scat, N_SPAT_, 16, sg0_W, scat, N_SPAT_, 16, sg0_W,
              ss_row, ss_col, ss_ea, E_SS_, sg0_as, sg0_ad, sg0_ae, sg0_We, sg0_b,
              HEADS_, CH_, 1, h1);
    gat_layer(st, B, G, h1, N_SPAT_, 64, sg1_W, h1, N_SPAT_, 64, sg1_W,
              ss_row, ss_col, ss_ea, E_SS_, sg1_as, sg1_ad, sg1_ae, sg1_We, sg1_b,
              HEADS_, CH_, 1, h2);

    // ---------------- C) bipartite spatial -> species ----------------
    pack_cols<<<gsz((long)N_SP_ * 64,  256), 256, 0, st>>>(part, 192, 0,  sp_gen,   64,  N_SP_);
    pack_cols<<<gsz((long)N_SP_ * 128, 256), 256, 0, st>>>(part, 192, 64, sp_phylo, 128, N_SP_);
    const int* b_row = bip_ei;
    const int* b_col = bip_ei + E_BIP_;
    gat_layer(st, B, G, h2, N_SPAT_, 64, bp_Ws, part, N_SP_, 192, bp_Wd,
              b_row, b_col, bip_ea, E_BIP_, bp_as, bp_ad, bp_ae, bp_We, bp_b,
              1, 64, 0, sts);

    // ---------------- D) species GNN ----------------
    pack_cols<<<gsz((long)N_SP_ * 64, 256), 256, 0, st>>>(xcat, 128, 0,  sts,  64, N_SP_);
    pack_cols<<<gsz((long)N_SP_ * 64, 256), 256, 0, st>>>(xcat, 128, 64, spin, 64, N_SP_);
    const int* p_row = sp_ei;
    const int* p_col = sp_ei + E_SP_;
    gat_layer(st, B, G, xcat, N_SP_, 128, pg0_W, xcat, N_SP_, 128, pg0_W,
              p_row, p_col, sp_ea, E_SP_, pg0_as, pg0_ad, pg0_ae, pg0_We, pg0_b,
              HEADS_, CH_, 1, x1);
    gat_layer(st, B, G, x1, N_SP_, 64, pg1_W, x1, N_SP_, 64, pg1_W,
              p_row, p_col, sp_ea, E_SP_, pg1_as, pg1_ad, pg1_ae, pg1_We, pg1_b,
              HEADS_, CH_, 1, x2);

    // ---------------- E) output head ----------------
    gemm(st, G, x2, fc_W, fc_b, msb, N_SP_, 64, 64, 0);
    split_softplus<<<gsz((long)N_SP_ * 64, 256), 256, 0, st>>>(msb, (float*)d_out, N_SP_);
}
